// _ModifiedRijalEtAl_58076547776650
// MI455X (gfx1250) — matrix operand order repaired and checked by reading, not compiled, not run on hardware
//
#include <hip/hip_runtime.h>

#define SEQ 2048
#define DIM 128
#define NB  32
#define NLAYERS 3

typedef __attribute__((ext_vector_type(16))) __bf16 v16bf;
typedef __attribute__((ext_vector_type(8)))  __bf16 v8bf;
typedef __attribute__((ext_vector_type(8)))  float  v8f;
typedef __attribute__((ext_vector_type(4)))  unsigned int v4u;
typedef __attribute__((ext_vector_type(8)))  int v8i;
typedef __attribute__((ext_vector_type(4)))  int v4i;

__device__ inline v16bf join8(v8bf lo, v8bf hi) {
  v16bf r;
#pragma unroll
  for (int i = 0; i < 8; ++i) { r[i] = lo[i]; r[i + 8] = hi[i]; }
  return r;
}

// ---------------------------------------------------------------------------
// Tensor Data Mover: 2D tile (tile_w x tile_h elements of 2 bytes) from global
// (row stride = stride_elems) into LDS at lds_addr, with LDS row padding
// (pad_int: 3 -> every 16 DWORDs, 5 -> every 64 DWORDs; pad_amt 3 -> 4 DWORDs).
// D# layout per CDNA5 ISA ch.8 (group0 128b, group1 256b, groups2/3 unused 2D).
// ---------------------------------------------------------------------------
__device__ inline void tdm_load_2d(unsigned lds_addr, const void* gptr,
                                   unsigned tile_w, unsigned tile_h,
                                   unsigned stride_elems,
                                   unsigned pad_int, unsigned pad_amt) {
  unsigned long long ga = (unsigned long long)gptr;
  v4u g0;
  g0[0] = 1u;                                      // count=1, user mode, no gather
  g0[1] = lds_addr;                                // LDS byte address
  g0[2] = (unsigned)ga;                            // global_addr[31:0]
  g0[3] = (unsigned)((ga >> 32) & 0x1ffffffull)    // global_addr[56:32]
          | (2u << 30);                            // type=2 ("image")
  v8i g1;
  g1[0] = (int)((1u << 16)                         // data_size = 2 bytes
                | (1u << 20)                       // pad_enable
                | (pad_int << 22) | (pad_amt << 25));
  g1[1] = (int)((stride_elems & 0xffffu) << 16);   // tensor_dim0[15:0] (use row stride)
  g1[2] = (int)(stride_elems >> 16);               // tensor_dim0[31:16]; tensor_dim1 lo16=0
  g1[3] = (int)(1u | (tile_w << 16));              // tensor_dim1 hi16 -> dim1=65536; tile_dim0
  g1[4] = (int)(tile_h & 0xffffu);                 // tile_dim1; tile_dim2=0
  g1[5] = (int)stride_elems;                       // tensor_dim0_stride[31:0]
  g1[6] = 0;                                       // stride hi16, tensor_dim1_stride lo16
  g1[7] = 0;
  v4i z4 = {0, 0, 0, 0};
  v8i z8 = {0, 0, 0, 0, 0, 0, 0, 0};
  __builtin_amdgcn_tensor_load_to_lds(g0, g1, z4, z4, z8, 0);
}

// ---------------------------------------------------------------------------
// x[b,l,d] = genotypes[b,l] * emb[l,d]   (f32)
// ---------------------------------------------------------------------------
__global__ __launch_bounds__(256) void embed_kernel(const float* __restrict__ g,
                                                    const float* __restrict__ emb,
                                                    float* __restrict__ x) {
  size_t i = ((size_t)blockIdx.x * blockDim.x + threadIdx.x) * 4;
  float gv = g[i >> 7];
  const float4 e = *(const float4*)(emb + (i & (size_t)(SEQ * DIM - 1)));
  float4 o; o.x = gv * e.x; o.y = gv * e.y; o.z = gv * e.z; o.w = gv * e.w;
  *(float4*)(x + i) = o;
}

// ---------------------------------------------------------------------------
// f32 -> bf16 weight conversion (all 3 layers of qw/kw/vw)
// ---------------------------------------------------------------------------
__global__ __launch_bounds__(256) void wconv_kernel(const float* __restrict__ qw,
                                                    const float* __restrict__ kw,
                                                    const float* __restrict__ vw,
                                                    __bf16* __restrict__ wq,
                                                    __bf16* __restrict__ wk,
                                                    __bf16* __restrict__ wv, int n) {
  int i = blockIdx.x * blockDim.x + threadIdx.x;
  if (i < n) {
    wq[i] = (__bf16)qw[i];
    wk[i] = (__bf16)kw[i];
    wv[i] = (__bf16)vw[i];
  }
}

// ---------------------------------------------------------------------------
// QKV projection: y = x @ W^T + b, output bf16.
// q/k stored row-major (B*L, D); v stored TRANSPOSED as vT[b][d][l] so the
// attention kernel's P@V B-operand is a contiguous-key 2D tile for the TDM.
// ---------------------------------------------------------------------------
__global__ __launch_bounds__(256) void qkv_kernel(
    const float* __restrict__ x,
    const __bf16* __restrict__ wq, const __bf16* __restrict__ wk, const __bf16* __restrict__ wv,
    const float* __restrict__ qb, const float* __restrict__ kb, const float* __restrict__ vb,
    __bf16* __restrict__ qo, __bf16* __restrict__ ko, __bf16* __restrict__ vo,
    int layer) {
  __shared__ __bf16 xl[16 * 136] __attribute__((aligned(16)));

  const int tid = threadIdx.x;
  const int wave = tid >> 5, lane = tid & 31;
  const int lh = lane & 15, hi = lane >> 4;
  const int sel = blockIdx.y;
  const size_t m0 = (size_t)blockIdx.x * 16;

  {  // stage 16x128 x-tile to LDS as bf16
    int e = tid * 8;
    int r = e >> 7, c = e & 127;
    const float* src = x + (m0 + r) * DIM + c;
    v8bf d;
#pragma unroll
    for (int i = 0; i < 8; ++i) d[i] = (__bf16)src[i];
    *(v8bf*)(&xl[r * 136 + c]) = d;
  }
  __syncthreads();

  const __bf16* W = (sel == 0 ? wq : sel == 1 ? wk : wv) + (size_t)layer * DIM * DIM;
  const float* bias = (sel == 0 ? qb : sel == 1 ? kb : vb) + (size_t)layer * DIM;

  const int e0 = wave * 16;
  v8f acc = {};
#pragma unroll
  for (int c = 0; c < 4; ++c) {
    int ca = c * 32 + hi * 8;
    v16bf A = join8(*(const v8bf*)(&xl[lh * 136 + ca]),
                    *(const v8bf*)(&xl[lh * 136 + ca + 16]));
    const __bf16* wrow = W + (size_t)(e0 + lh) * DIM + c * 32 + hi * 8;
    v16bf Bm = join8(*(const v8bf*)(wrow), *(const v8bf*)(wrow + 16));
    acc = __builtin_amdgcn_wmma_f32_16x16x32_bf16(false, A, false, Bm, (short)0, acc,
                                                  false, false);
  }
  float bv = bias[e0 + lh];

  if (sel == 2) {
    // transposed store: vT[(b*DIM + e)][l], 8 contiguous l per lane -> b128
    size_t bb = m0 >> 11;               // / SEQ
    size_t l0 = m0 & (SEQ - 1);
    v8bf pk;
#pragma unroll
    for (int r = 0; r < 8; ++r) pk[r] = (__bf16)(acc[r] + bv);
    __bf16* op = vo + (bb * DIM + e0 + lh) * SEQ + l0 + hi * 8;
    *(v8bf*)op = pk;
  } else {
    __bf16* obase = (sel == 0 ? qo : ko) + m0 * DIM;
#pragma unroll
    for (int r = 0; r < 8; ++r) {
      int row = r + hi * 8;
      obase[(size_t)row * DIM + e0 + lh] = (__bf16)(acc[r] + bv);
    }
  }
}

// ---------------------------------------------------------------------------
// Flash attention: 8 waves x 16 query rows per block, 32-key chunks.
// K and V^T chunks are staged to LDS by the Tensor Data Mover with double
// buffering (issue chunk i+1, compute WMMAs on chunk i, s_wait_tensorcnt).
// ---------------------------------------------------------------------------
__global__ __launch_bounds__(256) void attn_kernel(const __bf16* __restrict__ qg,
                                                   const __bf16* __restrict__ kg,
                                                   const __bf16* __restrict__ vtg,
                                                   float* __restrict__ xo) {
  __shared__ __bf16 kl[2 * 32 * 136] __attribute__((aligned(16)));  // K chunks (pad 4DW/row)
  __shared__ __bf16 vt[2 * 128 * 40] __attribute__((aligned(16)));  // V^T chunks (pad 4DW/row)
  __shared__ __bf16 pl[8 * 16 * 40]  __attribute__((aligned(16)));  // per-wave P scratch

  const int tid = threadIdx.x;
  const int wave = tid >> 5, lane = tid & 31;
  const int lh = lane & 15, hi = lane >> 4;
  const int b = blockIdx.y;
  const int q0 = blockIdx.x * 128;
  const int qrow = q0 + wave * 16 + lh;

  // preload this wave's 16x128 Q tile into A-operand registers
  const __bf16* qptr = qg + ((size_t)b * SEQ + qrow) * DIM;
  v16bf qa[4];
#pragma unroll
  for (int c = 0; c < 4; ++c) {
    int c0 = c * 32 + hi * 8;
    qa[c] = join8(*(const v8bf*)(qptr + c0), *(const v8bf*)(qptr + c0 + 16));
  }

  v8f acc[8];
  const v8f vz = {};
#pragma unroll
  for (int t = 0; t < 8; ++t) acc[t] = vz;
  float m[8], l[8];
#pragma unroll
  for (int r = 0; r < 8; ++r) { m[r] = -3.0e38f; l[r] = 0.f; }

  const __bf16* kbase = kg + (size_t)b * SEQ * DIM;   // row-major (l, d)
  const __bf16* vtb   = vtg + (size_t)b * DIM * SEQ;  // transposed (d, l)
  const unsigned kl_lds = (unsigned)(size_t)(void*)kl;
  const unsigned vt_lds = (unsigned)(size_t)(void*)vt;

  // prologue: TDM loads for chunk 0 into buffer 0 (one wave issues; EXEC ignored)
  if (wave == 0) {
    tdm_load_2d(kl_lds, kbase, DIM, 32, DIM, 5, 3);       // 32x128, LDS stride 272B
    tdm_load_2d(vt_lds, vtb, 32, DIM, SEQ, 3, 3);         // 128x32, LDS stride 80B
  }

  for (int kc = 0; kc < SEQ; kc += 32) {
    const int buf = (kc >> 5) & 1;
    if (wave == 0) __builtin_amdgcn_s_wait_tensorcnt(0);  // chunk kc has landed
    __syncthreads();
    if (wave == 0 && kc + 32 < SEQ) {                     // prefetch chunk kc+32
      tdm_load_2d(kl_lds + (unsigned)((buf ^ 1) * 32 * 136 * 2),
                  kbase + (size_t)(kc + 32) * DIM, DIM, 32, DIM, 5, 3);
      tdm_load_2d(vt_lds + (unsigned)((buf ^ 1) * 128 * 40 * 2),
                  vtb + (kc + 32), 32, DIM, SEQ, 3, 3);
    }
    const __bf16* klc = kl + buf * 32 * 136;
    const __bf16* vtc = vt + buf * 128 * 40;

    // ---- scores: two 16x16 tiles covering 32 keys ----
    v8f s0 = vz, s1 = vz;
#pragma unroll
    for (int c = 0; c < 4; ++c) {
      int col = c * 32 + hi * 8;
      v16bf b0 = join8(*(const v8bf*)(&klc[lh * 136 + col]),
                       *(const v8bf*)(&klc[lh * 136 + col + 16]));
      s0 = __builtin_amdgcn_wmma_f32_16x16x32_bf16(false, qa[c], false, b0, (short)0, s0,
                                                   false, false);
      v16bf b1 = join8(*(const v8bf*)(&klc[(16 + lh) * 136 + col]),
                       *(const v8bf*)(&klc[(16 + lh) * 136 + col + 16]));
      s1 = __builtin_amdgcn_wmma_f32_16x16x32_bf16(false, qa[c], false, b1, (short)0, s1,
                                                   false, false);
    }

    // ---- online softmax in C/D layout (row r(+8) across 16 lanes) ----
    float scale[8];
#pragma unroll
    for (int r = 0; r < 8; ++r) {
      float rm = fmaxf(s0[r], s1[r]);
#pragma unroll
      for (int off = 1; off < 16; off <<= 1) rm = fmaxf(rm, __shfl_xor(rm, off, 16));
      float mn = fmaxf(m[r], rm);
      float p0 = __expf(s0[r] - mn);
      float p1 = __expf(s1[r] - mn);
      float rs = p0 + p1;
#pragma unroll
      for (int off = 1; off < 16; off <<= 1) rs += __shfl_xor(rs, off, 16);
      float corr = __expf(m[r] - mn);
      l[r] = l[r] * corr + rs;
      m[r] = mn;
      scale[r] = corr;
      s0[r] = p0; s1[r] = p1;
    }
#pragma unroll
    for (int t = 0; t < 8; ++t)
#pragma unroll
      for (int r = 0; r < 8; ++r) acc[t][r] *= scale[r];

    // ---- re-layout P (D-layout -> A-layout) through per-wave LDS ----
    __bf16* pw = &pl[wave * 16 * 40];
#pragma unroll
    for (int r = 0; r < 8; ++r) {
      pw[(r + hi * 8) * 40 + lh]      = (__bf16)s0[r];
      pw[(r + hi * 8) * 40 + 16 + lh] = (__bf16)s1[r];
    }
    asm volatile("s_wait_dscnt 0" ::: "memory");
    int pc = hi * 8;
    v16bf pa = join8(*(const v8bf*)(&pw[lh * 40 + pc]),
                     *(const v8bf*)(&pw[lh * 40 + pc + 16]));

    // ---- acc += P @ V over 8 d-tiles (Bt[n][k] = vT[d][key]) ----
#pragma unroll
    for (int t = 0; t < 8; ++t) {
      int drow = t * 16 + lh;
      v16bf bv = join8(*(const v8bf*)(&vtc[drow * 40 + hi * 8]),
                       *(const v8bf*)(&vtc[drow * 40 + hi * 8 + 16]));
      acc[t] = __builtin_amdgcn_wmma_f32_16x16x32_bf16(false, pa, false, bv, (short)0,
                                                       acc[t], false, false);
    }
  }

  // epilogue: normalize and store f32 (overwrites layer input x)
  float* xrow = xo + ((size_t)b * SEQ + q0 + wave * 16) * DIM;
#pragma unroll
  for (int t = 0; t < 8; ++t)
#pragma unroll
    for (int r = 0; r < 8; ++r) {
      int row = r + hi * 8;
      xrow[(size_t)row * DIM + t * 16 + lh] = acc[t][r] / l[r];
    }
}

// ---------------------------------------------------------------------------
// out[b] = dot(x[b, :], rw) + rb
// ---------------------------------------------------------------------------
__global__ __launch_bounds__(256) void readout_kernel(const float* __restrict__ x,
                                                      const float* __restrict__ rw,
                                                      const float* __restrict__ rb,
                                                      float* __restrict__ out) {
  __shared__ float red[256];
  const int b = blockIdx.x, tid = threadIdx.x;
  const float* xb = x + (size_t)b * SEQ * DIM;
  float s = 0.f;
  for (size_t i = (size_t)tid * 4; i < (size_t)SEQ * DIM; i += 256 * 4) {
    float4 a = *(const float4*)(xb + i);
    float4 w = *(const float4*)(rw + i);
    s += a.x * w.x + a.y * w.y + a.z * w.z + a.w * w.w;
  }
  red[tid] = s;
  __syncthreads();
  for (int st = 128; st > 0; st >>= 1) {
    if (tid < st) red[tid] += red[tid + st];
    __syncthreads();
  }
  if (tid == 0) out[b] = red[0] + rb[0];
}

// ---------------------------------------------------------------------------
extern "C" void kernel_launch(void* const* d_in, const int* in_sizes, int n_in,
                              void* d_out, int out_size, void* d_ws, size_t ws_size,
                              hipStream_t stream) {
  (void)in_sizes; (void)n_in; (void)out_size; (void)ws_size;
  const float* g   = (const float*)d_in[0];
  const float* emb = (const float*)d_in[1];
  const float* qw  = (const float*)d_in[2];
  const float* qb  = (const float*)d_in[3];
  const float* kw  = (const float*)d_in[4];
  const float* kb  = (const float*)d_in[5];
  const float* vw  = (const float*)d_in[6];
  const float* vb  = (const float*)d_in[7];
  const float* rw  = (const float*)d_in[8];
  const float* rb  = (const float*)d_in[9];
  float* out = (float*)d_out;

  // workspace layout (bytes)
  char* ws = (char*)d_ws;
  float*  x   = (float*)(ws + 0);             // 32 MB f32 activations
  __bf16* q   = (__bf16*)(ws + 33554432);     // 16 MB, row-major
  __bf16* k   = (__bf16*)(ws + 50331648);     // 16 MB, row-major
  __bf16* vT  = (__bf16*)(ws + 67108864);     // 16 MB, transposed (b, d, l)
  __bf16* wqb = (__bf16*)(ws + 83886080);     // 96 KB each
  __bf16* wkb = (__bf16*)(ws + 83984384);
  __bf16* wvb = (__bf16*)(ws + 84082688);

  wconv_kernel<<<192, 256, 0, stream>>>(qw, kw, vw, wqb, wkb, wvb,
                                        NLAYERS * DIM * DIM);
  embed_kernel<<<(NB * SEQ * DIM) / (4 * 256), 256, 0, stream>>>(g, emb, x);
  for (int layer = 0; layer < NLAYERS; ++layer) {
    qkv_kernel<<<dim3((NB * SEQ) / 16, 3), 256, 0, stream>>>(
        x, wqb, wkb, wvb, qb, kb, vb, q, k, vT, layer);
    attn_kernel<<<dim3(SEQ / 128, NB), 256, 0, stream>>>(q, k, vT, x);
  }
  readout_kernel<<<NB, 256, 0, stream>>>(x, rw, rb, out);
}
